// SOMLayer_67542655697078
// MI455X (gfx1250) — compile-verified, matrix operand order checked by reading, not run on hardware
//
#include <hip/hip_runtime.h>

typedef __bf16 bf16_t;
typedef __attribute__((ext_vector_type(16))) __bf16 v16bf;
typedef __attribute__((ext_vector_type(8)))  __bf16 v8bf;
typedef __attribute__((ext_vector_type(8)))  float  v8f;
typedef __attribute__((ext_vector_type(4)))  float  v4f;

#define N_ROWS  32768
#define M_COLS  4096
#define DDIM    512
#define RPB     128        // rows per block (8 waves x 16)
#define THREADS 256
#define BSTRIDE 520        // padded bf16 row stride: 1040B = 65*16B aligned, 4-bank rotation
#define BBUF    (16 * BSTRIDE)
#define KCHUNKS 16         // 512 / 32
#define CTILES  256        // 4096 / 16

__global__ __launch_bounds__(THREADS, 1)
void som_winner_kernel(const float* __restrict__ z,
                       const float* __restrict__ protos,
                       float* __restrict__ out)
{
    __shared__ __align__(16) bf16_t sBhi[2 * BBUF];   // 33,280 B (double-buffered)
    __shared__ __align__(16) bf16_t sBlo[2 * BBUF];   // 33,280 B
    __shared__ float sZ2[RPB];
    __shared__ float sZpart[THREADS];
    __shared__ float sPpart[2][THREADS];

    const int tid   = threadIdx.x;
    const int wave  = tid >> 5;
    const int lane  = tid & 31;
    const int row0  = blockIdx.x * RPB;
    const int m_row = lane & 15;   // A-row within wave tile / B-col (C/D N index)
    const int khalf = lane >> 4;   // K-half select per ISA fragment layout

    // ---------- Phase A: load this wave's 16 z-rows straight into registers as
    // split bf16 (hi+lo) in WMMA A-fragment layout; accumulate ||z||^2 partials.
    v16bf aHi[KCHUNKS];
    v16bf aLo[KCHUNKS];
    {
        const float* rowp = z + (size_t)(row0 + wave * 16 + m_row) * DDIM + khalf * 8;
        float a2 = 0.f;
        #pragma unroll
        for (int c = 0; c < KCHUNKS; ++c) {
            const float* p = rowp + c * 32;
            v4f f0 = *(const v4f*)(p);        // K = base+0..3
            v4f f1 = *(const v4f*)(p + 4);    // K = base+4..7
            v4f f2 = *(const v4f*)(p + 16);   // K = base+16..19
            v4f f3 = *(const v4f*)(p + 20);   // K = base+20..23
            #pragma unroll
            for (int j = 0; j < 4; ++j) {
                float x0 = f0[j], x1 = f1[j], x2 = f2[j], x3 = f3[j];
                bf16_t h0 = (bf16_t)x0, h1 = (bf16_t)x1, h2 = (bf16_t)x2, h3 = (bf16_t)x3;
                aHi[c][j]      = h0;  aLo[c][j]      = (bf16_t)(x0 - (float)h0);
                aHi[c][4 + j]  = h1;  aLo[c][4 + j]  = (bf16_t)(x1 - (float)h1);
                aHi[c][8 + j]  = h2;  aLo[c][8 + j]  = (bf16_t)(x2 - (float)h2);
                aHi[c][12 + j] = h3;  aLo[c][12 + j] = (bf16_t)(x3 - (float)h3);
                a2 += x0 * x0 + x1 * x1 + x2 * x2 + x3 * x3;
            }
        }
        sZpart[tid] = a2;
    }
    __syncthreads();
    if (tid < RPB) {
        const int w2 = tid >> 4, r = tid & 15;
        sZ2[tid] = sZpart[w2 * 32 + r] + sZpart[w2 * 32 + 16 + r];
    }

    // ---------- Pipelined main loop over prototype column tiles (16 cols each)
    float bestD[8];
    int   bestI[8];
    #pragma unroll
    for (int v = 0; v < 8; ++v) { bestD[v] = 3.4e38f; bestI[v] = 0; }

    const bf16_t* bHiB = sBhi + m_row * BSTRIDE + khalf * 8;
    const bf16_t* bLoB = sBlo + m_row * BSTRIDE + khalf * 8;

    const int spr = tid >> 4;           // staging: prototype row 0..15
    const int sdb = (tid & 15) * 32;    // staging: dim base
    bf16_t* const dhi0 = sBhi + spr * BSTRIDE + sdb;
    bf16_t* const dlo0 = sBlo + spr * BSTRIDE + sdb;

    // Prologue: load + convert + store tile 0 into buffer 0
    {
        const float* src = protos + (size_t)spr * DDIM + sdb;
        float p2a = 0.f;
        #pragma unroll
        for (int i = 0; i < 8; ++i) {
            v4f v = *(const v4f*)(src + i * 4);
            #pragma unroll
            for (int j = 0; j < 4; ++j) {
                float f = v[j];
                bf16_t h = (bf16_t)f;
                dhi0[i * 4 + j] = h;
                dlo0[i * 4 + j] = (bf16_t)(f - (float)h);
                p2a += f * f;
            }
        }
        sPpart[0][tid] = p2a;
    }
    __syncthreads();

    #pragma unroll 1
    for (int t = 0; t < CTILES; ++t) {
        const int cur = t & 1;
        const int nxt = cur ^ 1;

        // Issue global loads for tile t+1 now; they complete under the WMMA loop.
        v4f stg[8];
        if (t + 1 < CTILES) {
            const float* src = protos + (size_t)((t + 1) * 16 + spr) * DDIM + sdb;
            #pragma unroll
            for (int i = 0; i < 8; ++i) stg[i] = *(const v4f*)(src + i * 4);
            if (t + 2 < CTILES) {   // L2 insurance for the tile after that
                __builtin_prefetch(protos + (size_t)((t + 2) * 16 + spr) * DDIM + sdb, 0, 1);
            }
        }

        // ---- Compute tile t from buffer `cur`
        v8f acc = {};
        #pragma unroll
        for (int c = 0; c < KCHUNKS; ++c) {
            const int off = cur * BBUF + c * 32;
            v8bf bh0 = *(const v8bf*)(bHiB + off);
            v8bf bh1 = *(const v8bf*)(bHiB + off + 16);
            v8bf bl0 = *(const v8bf*)(bLoB + off);
            v8bf bl1 = *(const v8bf*)(bLoB + off + 16);
            v16bf bh = __builtin_shufflevector(bh0, bh1, 0,1,2,3,4,5,6,7,8,9,10,11,12,13,14,15);
            v16bf bl = __builtin_shufflevector(bl0, bl1, 0,1,2,3,4,5,6,7,8,9,10,11,12,13,14,15);
            // split-bf16 fp32 emulation: hi*hi + hi*lo + lo*hi
            acc = __builtin_amdgcn_wmma_f32_16x16x32_bf16(false, aHi[c], false, bh, (short)0, acc, false, false);
            acc = __builtin_amdgcn_wmma_f32_16x16x32_bf16(false, aHi[c], false, bl, (short)0, acc, false, false);
            acc = __builtin_amdgcn_wmma_f32_16x16x32_bf16(false, aLo[c], false, bh, (short)0, acc, false, false);
        }

        float p2 = 0.f;
        #pragma unroll
        for (int i = 0; i < 16; ++i) p2 += sPpart[cur][m_row * 16 + i];

        const int colg = t * 16 + m_row;
        #pragma unroll
        for (int v = 0; v < 8; ++v) {
            const int rl = v + 8 * khalf;  // C/D layout: VGPR v holds M = v (+8 for upper half)
            float d2 = sZ2[wave * 16 + rl] - 2.f * acc[v] + p2;
            if (d2 < bestD[v]) { bestD[v] = d2; bestI[v] = colg; }
        }

        // ---- Convert + store tile t+1 into buffer `nxt` (safe: compute(t) used `cur`;
        // buffer `nxt` was last read in iteration t-1, before the previous barrier)
        if (t + 1 < CTILES) {
            bf16_t* dhi = dhi0 + nxt * BBUF;
            bf16_t* dlo = dlo0 + nxt * BBUF;
            float p2a = 0.f;
            #pragma unroll
            for (int i = 0; i < 8; ++i) {
                #pragma unroll
                for (int j = 0; j < 4; ++j) {
                    float f = stg[i][j];
                    bf16_t h = (bf16_t)f;
                    dhi[i * 4 + j] = h;
                    dlo[i * 4 + j] = (bf16_t)(f - (float)h);
                    p2a += f * f;
                }
            }
            sPpart[nxt][tid] = p2a;
        }
        __syncthreads();   // single barrier per tile
    }

    // ---------- Cross-lane argmin reduction (reuse B buffers as scratch)
    float* redD = (float*)sBhi;   // 128*16*4 = 8192 B, fits
    int*   redI = (int*)sBlo;
    #pragma unroll
    for (int v = 0; v < 8; ++v) {
        const int rl = v + 8 * khalf;
        redD[(wave * 16 + rl) * 16 + m_row] = bestD[v];
        redI[(wave * 16 + rl) * 16 + m_row] = bestI[v];
    }
    __syncthreads();
    int* sBest = (int*)sZpart;
    if (tid < RPB) {
        float bd = redD[tid * 16];
        int   bi = redI[tid * 16];
        for (int c = 1; c < 16; ++c) {
            float d = redD[tid * 16 + c];
            int  ii = redI[tid * 16 + c];
            if (d < bd || (d == bd && ii < bi)) { bd = d; bi = ii; }
        }
        sBest[tid] = bi;
        out[(size_t)N_ROWS * DDIM + row0 + tid] = (float)bi;  // min_idx output
    }
    __syncthreads();

    // ---------- Gather winner prototypes
    {
        const int r  = tid >> 1;
        const int hf = tid & 1;
        const int wi = sBest[r];
        const float* src = protos + (size_t)wi * DDIM + hf * 256;
        float* dst = out + (size_t)(row0 + r) * DDIM + hf * 256;
        #pragma unroll 4
        for (int i = 0; i < 64; ++i)
            *(v4f*)(dst + i * 4) = *(const v4f*)(src + i * 4);
    }
}

extern "C" void kernel_launch(void* const* d_in, const int* in_sizes, int n_in,
                              void* d_out, int out_size, void* d_ws, size_t ws_size,
                              hipStream_t stream) {
    (void)in_sizes; (void)n_in; (void)out_size; (void)d_ws; (void)ws_size;
    const float* z      = (const float*)d_in[0];
    const float* protos = (const float*)d_in[1];
    float* out = (float*)d_out;
    dim3 grid(N_ROWS / RPB);   // 256 workgroups
    dim3 block(THREADS);       // 8 wave32
    som_winner_kernel<<<grid, block, 0, stream>>>(z, protos, out);
}
